// ICGFitter_64604898066506
// MI455X (gfx1250) — matrix operand order repaired and checked by reading, not compiled, not run on hardware
//
#include <hip/hip_runtime.h>
#include <hip/hip_fp16.h>
#include <stdint.h>

// CDNA5 / gfx1250 — wave32, WMMA 16x16x32 f16->f32.
typedef __attribute__((ext_vector_type(16))) _Float16 v16h;
typedef __attribute__((ext_vector_type(8)))  float    v8f;

#define KC 64     // communities
#define FF 128    // features

union Frag16 { v16h h; unsigned int u[8]; };

__device__ __forceinline__ float sigmoidf_(float x) {
    return 1.0f / (1.0f + __expf(-x));
}

// ---------------------------------------------------------------------------
// init: transpose feat_mat to f16 [F][K], zero M accumulator, sums, and the
// zero-pad columns of the transposed A / As arrays.
// ---------------------------------------------------------------------------
__global__ void icg_init_kernel(const float* __restrict__ feat,
                                _Float16* __restrict__ feat_hT,
                                float* __restrict__ Mbuf,
                                float* __restrict__ sums,
                                _Float16* __restrict__ A_hT,
                                _Float16* __restrict__ As_hT,
                                int N, int Npad)
{
    int i = blockIdx.x * blockDim.x + threadIdx.x;
    int padN = Npad - N;
    if (i < KC * FF) {
        int k = i >> 7, f = i & (FF - 1);
        feat_hT[f * KC + k] = (_Float16)feat[k * FF + f];
    } else if (i < KC * FF + KC * KC) {
        Mbuf[i - KC * FF] = 0.0f;
    } else if (i < KC * FF + KC * KC + 2) {
        sums[i - KC * FF - KC * KC] = 0.0f;
    } else {
        int j = i - (KC * FF + KC * KC + 2);
        if (padN > 0) {
            if (j < KC * padN) {
                int k = j / padN, n = N + j % padN;
                A_hT[k * Npad + n] = (_Float16)0.0f;
            } else if (j < 2 * KC * padN) {
                j -= KC * padN;
                int k = j / padN, n = N + j % padN;
                As_hT[k * Npad + n] = (_Float16)0.0f;
            }
        }
    }
}

// ---------------------------------------------------------------------------
// prep: A = sigmoid(logits); write As row-major f16 and A / As transposed f16.
// ---------------------------------------------------------------------------
__global__ void icg_prep_kernel(const float* __restrict__ logits,
                                const float* __restrict__ scal,
                                _Float16* __restrict__ A_hT,
                                _Float16* __restrict__ As_hT,
                                _Float16* __restrict__ As_h,
                                int N, int Npad)
{
    int i = blockIdx.x * blockDim.x + threadIdx.x;
    if (i >= N * KC) return;
    int n = i >> 6, k = i & (KC - 1);
    float a  = sigmoidf_(logits[i]);
    float as = a * scal[k];
    As_h[i]              = (_Float16)as;
    A_hT[k * Npad + n]   = (_Float16)a;
    As_hT[k * Npad + n]  = (_Float16)as;
}

// ---------------------------------------------------------------------------
// M = A^T * As  (64x64, reduce over N) via v_wmma_f32_16x16x32_f16.
// Wave w (of 4 per block) owns k-rows [16w,16w+16); accumulates all 4 l-tiles.
// ---------------------------------------------------------------------------
__global__ void icg_wmma_M_kernel(const _Float16* __restrict__ A_hT,
                                  const _Float16* __restrict__ As_hT,
                                  float* __restrict__ Mbuf,
                                  int Npad)
{
    int lane = threadIdx.x & 31;
    int wave = threadIdx.x >> 5;            // ki tile = wave
    int half = lane >> 4;
    int lm   = lane & 15;
    int kRow = wave * 16 + lm;              // A-matrix row (k index of M)
    int nchunks = Npad >> 5;

    v8f acc0 = {}, acc1 = {}, acc2 = {}, acc3 = {};

    for (int c = blockIdx.x; c < nchunks; c += gridDim.x) {
        int nb = c << 5;
        Frag16 fa, fb0, fb1, fb2, fb3;
        const _Float16* aRow = A_hT + (size_t)kRow * Npad + nb;
#pragma unroll
        for (int v = 0; v < 8; ++v) {
            // A operand, this code's contiguous k order (same as B): VGPR v holds K-dim pair 2v + 16*half
            int nOffA = 2 * v + 16 * half;
            fa.u[v] = *(const unsigned int*)(aRow + nOffA);
        }
#pragma unroll
        for (int v = 0; v < 8; ++v) {
            // B operand, this code's contiguous k order (same as A): VGPR v holds K-dim pair 2v + 16*half, column lm
            int nOffB = 2 * v + 16 * half;
            fb0.u[v] = *(const unsigned int*)(As_hT + (size_t)(0 * 16 + lm) * Npad + nb + nOffB);
            fb1.u[v] = *(const unsigned int*)(As_hT + (size_t)(1 * 16 + lm) * Npad + nb + nOffB);
            fb2.u[v] = *(const unsigned int*)(As_hT + (size_t)(2 * 16 + lm) * Npad + nb + nOffB);
            fb3.u[v] = *(const unsigned int*)(As_hT + (size_t)(3 * 16 + lm) * Npad + nb + nOffB);
        }
        acc0 = __builtin_amdgcn_wmma_f32_16x16x32_f16(false, fa.h, false, fb0.h, (short)0, acc0, false, false);
        acc1 = __builtin_amdgcn_wmma_f32_16x16x32_f16(false, fa.h, false, fb1.h, (short)0, acc1, false, false);
        acc2 = __builtin_amdgcn_wmma_f32_16x16x32_f16(false, fa.h, false, fb2.h, (short)0, acc2, false, false);
        acc3 = __builtin_amdgcn_wmma_f32_16x16x32_f16(false, fa.h, false, fb3.h, (short)0, acc3, false, false);
    }

#pragma unroll
    for (int r = 0; r < 8; ++r) {
        int mm = wave * 16 + r + 8 * half;   // D layout: m = r + 8*(lane>=16)
        atomicAdd(&Mbuf[mm * KC +  0 + lm], acc0[r]);
        atomicAdd(&Mbuf[mm * KC + 16 + lm], acc1[r]);
        atomicAdd(&Mbuf[mm * KC + 32 + lm], acc2[r]);
        atomicAdd(&Mbuf[mm * KC + 48 + lm], acc3[r]);
    }
}

// ---------------------------------------------------------------------------
// feature loss: sum((x - As@feat)^2). One 16x16 output tile per wave per task.
// ---------------------------------------------------------------------------
__global__ void icg_wmma_feat_kernel(const _Float16* __restrict__ As_h,
                                     const _Float16* __restrict__ feat_hT,
                                     const float* __restrict__ x,
                                     float* __restrict__ feat_sum,
                                     int N)
{
    int lane = threadIdx.x & 31;
    int wIb  = threadIdx.x >> 5;
    int wPb  = blockDim.x >> 5;
    int gwave  = blockIdx.x * wPb + wIb;
    int nwaves = gridDim.x * wPb;
    int half = lane >> 4, lm = lane & 15;

    int tilesM = N >> 4;                 // 3125
    int T = tilesM * (FF >> 4);          // * 8 column tiles

    float ssq = 0.0f;
    for (int t = gwave; t < T; t += nwaves) {
        int tm = t >> 3, tf = t & 7;
        int rowN = tm * 16 + lm;         // A-matrix row (node)
        int colF = tf * 16 + lm;         // B-matrix column (feature)
        v8f acc = {};
#pragma unroll
        for (int kc = 0; kc < 2; ++kc) {
            Frag16 fa, fb;
#pragma unroll
            for (int v = 0; v < 8; ++v) {
                int kOffA = 2 * v + 16 * half + 32 * kc;
                fa.u[v] = *(const unsigned int*)(As_h + (size_t)rowN * KC + kOffA);
                int kOffB = 2 * v + 16 * half + 32 * kc;
                fb.u[v] = *(const unsigned int*)(feat_hT + (size_t)colF * KC + kOffB);
            }
            acc = __builtin_amdgcn_wmma_f32_16x16x32_f16(false, fa.h, false, fb.h, (short)0, acc, false, false);
        }
#pragma unroll
        for (int r = 0; r < 8; ++r) {
            int mm = tm * 16 + r + 8 * half;
            int ff = tf * 16 + lm;
            float d = x[(size_t)mm * FF + ff] - acc[r];
            ssq += d * d;
        }
    }

    __shared__ float red[128];
    red[threadIdx.x] = ssq;
    __syncthreads();
    for (int s = blockDim.x >> 1; s > 0; s >>= 1) {
        if (threadIdx.x < s) red[threadIdx.x] += red[threadIdx.x + s];
        __syncthreads();
    }
    if (threadIdx.x == 0) atomicAdd(feat_sum, red[0]);
}

// ---------------------------------------------------------------------------
// local term: sum over edges of sum_k sigmoid(l[dst,k])*s[k]*sigmoid(l[src,k])
// A (12.8 MB) is L2-resident; recompute sigmoid instead of storing f32 A.
// ---------------------------------------------------------------------------
__global__ void icg_local_kernel(const float* __restrict__ logits,
                                 const float* __restrict__ scal,
                                 const int* __restrict__ edges,
                                 float* __restrict__ local_sum,
                                 int E)
{
    int e = blockIdx.x * blockDim.x + threadIdx.x;
    float acc = 0.0f;
    if (e < E) {
        int s_ = edges[e];
        int d_ = edges[E + e];
        const float4* rs = (const float4*)(logits + (size_t)s_ * KC);
        const float4* rd = (const float4*)(logits + (size_t)d_ * KC);
        const float4* sc = (const float4*)scal;
#pragma unroll 4
        for (int q = 0; q < KC / 4; ++q) {
            float4 a = rs[q], b = rd[q], w = sc[q];
            acc += sigmoidf_(a.x) * w.x * sigmoidf_(b.x)
                 + sigmoidf_(a.y) * w.y * sigmoidf_(b.y)
                 + sigmoidf_(a.z) * w.z * sigmoidf_(b.z)
                 + sigmoidf_(a.w) * w.w * sigmoidf_(b.w);
        }
    }
    __shared__ float red[256];
    red[threadIdx.x] = acc;
    __syncthreads();
    for (int s = blockDim.x >> 1; s > 0; s >>= 1) {
        if (threadIdx.x < s) red[threadIdx.x] += red[threadIdx.x + s];
        __syncthreads();
    }
    if (threadIdx.x == 0) atomicAdd(local_sum, red[0]);
}

// ---------------------------------------------------------------------------
// final: global_term = sum(M * M^T); combine everything into scalar loss.
// ---------------------------------------------------------------------------
__global__ void icg_final_kernel(const float* __restrict__ Mbuf,
                                 const float* __restrict__ sums,
                                 float* __restrict__ out,
                                 int N, int E)
{
    float g = 0.0f;
    for (int i = threadIdx.x; i < KC * KC; i += blockDim.x) {
        int k = i >> 6, l = i & (KC - 1);
        g += Mbuf[i] * Mbuf[l * KC + k];
    }
    __shared__ float red[128];
    red[threadIdx.x] = g;
    __syncthreads();
    for (int s = blockDim.x >> 1; s > 0; s >>= 1) {
        if (threadIdx.x < s) red[threadIdx.x] += red[threadIdx.x + s];
        __syncthreads();
    }
    if (threadIdx.x == 0) {
        float global_term = red[0];
        float loss = (global_term - 2.0f * sums[0] + (float)E) / (float)N;
        loss += 0.1f * sums[1] / (float)FF;
        out[0] = loss;
    }
}

// ---------------------------------------------------------------------------
extern "C" void kernel_launch(void* const* d_in, const int* in_sizes, int n_in,
                              void* d_out, int out_size, void* d_ws, size_t ws_size,
                              hipStream_t stream)
{
    const float* x      = (const float*)d_in[0];   // [N, F]
    const int*   edges  = (const int*)  d_in[1];   // [2, E]
    const float* logits = (const float*)d_in[2];   // [N, K]
    const float* scal   = (const float*)d_in[3];   // [K]
    const float* feat   = (const float*)d_in[4];   // [K, F]
    float* out = (float*)d_out;

    const int N = in_sizes[0] / FF;
    const int E = in_sizes[1] / 2;
    const int Npad = (N + 31) & ~31;

    // workspace carve-out (256B aligned)
    char* ws = (char*)d_ws;
    size_t off = 0;
    auto carve = [&](size_t bytes) -> char* {
        char* p = ws + off;
        off += (bytes + 255) & ~(size_t)255;
        return p;
    };
    _Float16* A_hT    = (_Float16*)carve((size_t)KC * Npad * sizeof(_Float16)); // [K][Npad]
    _Float16* As_hT   = (_Float16*)carve((size_t)KC * Npad * sizeof(_Float16)); // [K][Npad]
    _Float16* As_h    = (_Float16*)carve((size_t)N * KC * sizeof(_Float16));    // [N][K]
    _Float16* feat_hT = (_Float16*)carve((size_t)FF * KC * sizeof(_Float16));   // [F][K]
    float*    Mbuf    = (float*)   carve((size_t)KC * KC * sizeof(float));      // [K][K]
    float*    sums    = (float*)   carve(2 * sizeof(float));  // [local_sum, feat_sum]
    (void)ws_size;

    // 1) init (feat transpose + zeroing)
    {
        int total = KC * FF + KC * KC + 2 + 2 * KC * (Npad - N);
        int blk = 256, grd = (total + blk - 1) / blk;
        icg_init_kernel<<<grd, blk, 0, stream>>>(feat, feat_hT, Mbuf, sums, A_hT, As_hT, N, Npad);
    }
    // 2) sigmoid + f16 conversion / transposition
    {
        int total = N * KC;
        int blk = 256, grd = (total + blk - 1) / blk;
        icg_prep_kernel<<<grd, blk, 0, stream>>>(logits, scal, A_hT, As_hT, As_h, N, Npad);
    }
    // 3) M = A^T As via WMMA
    icg_wmma_M_kernel<<<120, 128, 0, stream>>>(A_hT, As_hT, Mbuf, Npad);
    // 4) feature reconstruction loss via WMMA
    icg_wmma_feat_kernel<<<256, 128, 0, stream>>>(As_h, feat_hT, x, &sums[1], N);
    // 5) edge local term
    {
        int blk = 256, grd = (E + blk - 1) / blk;
        icg_local_kernel<<<grd, blk, 0, stream>>>(logits, scal, edges, &sums[0], E);
    }
    // 6) combine into scalar loss
    icg_final_kernel<<<1, 128, 0, stream>>>(Mbuf, sums, out, N, E);
}
